// RouterCNN_88768384073775
// MI455X (gfx1250) — matrix operand order repaired and checked by reading, not compiled, not run on hardware
//
#include <hip/hip_runtime.h>
#include <hip/hip_bf16.h>

typedef __attribute__((ext_vector_type(16))) _Float16 v16h;
typedef __attribute__((ext_vector_type(8)))  float    v8f;

#define BATCH 8192
#define HID 16
#define NEXP 6
#define SCALE 0.95f

union F16x16 { uint4 q[2]; v16h v; _Float16 h[16]; };
union F16x8  { uint4 q;    _Float16 h[8]; };

// ---------------------------------------------------------------------------
// Weight pre-swizzle: expert_w [6][16][16][3][3] f32 -> A-fragment layout f16.
// K = tap*16 + ch (tap = dy*3+dx), padded 144 -> 160 (5 chunks of 32).
// A 16x32 k order chosen here (B uses the same): lane L (half = L>>4, M = L&15), element j in v16h:
//   K_local = j + 16*half;  K >= 144 -> 0 (nulls the tap-9 rows).
// ---------------------------------------------------------------------------
__global__ void prep_wfrag_kernel(const float* __restrict__ ew,
                                  _Float16* __restrict__ wfrag) {
    int tid = blockIdx.x * blockDim.x + threadIdx.x;
    if (tid >= NEXP * 5 * 32) return;
    int lane = tid & 31;
    int kc   = (tid >> 5) % 5;
    int e    = tid / (5 * 32);
    int half = lane >> 4;
    int M    = lane & 15;
    _Float16* dst = wfrag + (size_t)tid * 16;
#pragma unroll
    for (int j = 0; j < 16; ++j) {
        int Klocal = j + 16 * half;
        int K = kc * 32 + Klocal;
        float v = 0.0f;
        if (K < 144) {
            int tap = K >> 4;   // 0..8
            int ch  = K & 15;
            v = ew[((e * HID + M) * HID + ch) * 9 + tap];
        }
        dst[j] = (_Float16)v;
    }
}

// ---------------------------------------------------------------------------
// Gate matrix -> f16, reordered to h element order (elem = p*16+c), padded to
// 16 "expert" rows (rows 6..15 = 0) so WMMA B-fragment loads are branchless.
// ---------------------------------------------------------------------------
__global__ void prep_gate_kernel(const float* __restrict__ gw,
                                 _Float16* __restrict__ gf) {
    int i = blockIdx.x * blockDim.x + threadIdx.x;   // 16*4096
    if (i >= 16 * 4096) return;
    int e = i >> 12, elem = i & 4095;
    int p = elem >> 4, c = elem & 15;
    float v = (e < NEXP) ? gw[e * 4096 + c * 256 + p] : 0.0f;
    gf[i] = (_Float16)v;
}

// ---------------------------------------------------------------------------
// First conv 3->16, 3x3, stride 2, pad 1, relu. x [B,3,32,32] f32.
// Output h0 f16, layout [b][p=oy*16+ox][c].  One block per sample.
// ---------------------------------------------------------------------------
__global__ __launch_bounds__(256)
void first_conv_kernel(const float* __restrict__ x,
                       const float* __restrict__ fw,
                       const float* __restrict__ fb,
                       _Float16* __restrict__ h0) {
    __shared__ float w[432];
    __shared__ float bia[16];
    for (int i = threadIdx.x; i < 432; i += 256) w[i] = fw[i];
    if (threadIdx.x < 16) bia[threadIdx.x] = fb[threadIdx.x];
    __syncthreads();

    const int b = blockIdx.x;
    const int p = threadIdx.x;           // 0..255
    const int oy = p >> 4, ox = p & 15;

    float acc[16];
#pragma unroll
    for (int c = 0; c < 16; ++c) acc[c] = bia[c];

    for (int ch = 0; ch < 3; ++ch) {
#pragma unroll
        for (int ky = 0; ky < 3; ++ky) {
            int iy = 2 * oy + ky - 1;
            if ((unsigned)iy >= 32u) continue;
#pragma unroll
            for (int kx = 0; kx < 3; ++kx) {
                int ix = 2 * ox + kx - 1;
                if ((unsigned)ix >= 32u) continue;
                float xv = x[((size_t)(b * 3 + ch) << 10) + (iy << 5) + ix];
#pragma unroll
                for (int c = 0; c < 16; ++c)
                    acc[c] = fmaf(xv, w[((c * 3 + ch) * 3 + ky) * 3 + kx], acc[c]);
            }
        }
    }
    F16x16 o;
#pragma unroll
    for (int c = 0; c < 16; ++c) o.h[c] = (_Float16)fmaxf(acc[c], 0.0f);
    uint4* dst = (uint4*)(h0 + ((size_t)b * 4096 + (size_t)p * 16));
    dst[0] = o.q[0];
    dst[1] = o.q[1];
}

// ---------------------------------------------------------------------------
// Gate via WMMA: logits = h [B x 4096] @ gate_f16^T [4096 x 16(6 used)].
// One wave per 16-sample tile, 8 waves/block -> 128 samples per block.
// A fragment: lane (half,M): sample = tile+M, two contiguous b128 from h.
// B fragment: lane column e = L%16, two b128 of gate_f16[e] at the same k as A.
// K = 4096 -> 128 chunks of 32.  Argmax (softmax monotone, top-1 w == 0.95).
// ---------------------------------------------------------------------------
__global__ __launch_bounds__(256)
void gate_kernel(const _Float16* __restrict__ hin,
                 const _Float16* __restrict__ gf,
                 int* __restrict__ idx) {
    __shared__ float logits[8 * 16 * 8];   // [wave][sample][expert(8)]

    const int wave = threadIdx.x >> 5;
    const int lane = threadIdx.x & 31;
    const int half = lane >> 4;
    const int col  = lane & 15;            // A: sample row M; B: expert column
    const int tile = blockIdx.x * 8 + wave;      // 16 samples per tile
    const int samp = tile * 16 + col;

    const uint4* ha = (const uint4*)(hin + (size_t)samp * 4096);  // 512 uint4
    const uint4* gb = (const uint4*)gf;
    __builtin_prefetch(ha, 0, 3);          // global_prefetch_b8 (gfx1250)

    v8f c;
#pragma unroll
    for (int v = 0; v < 8; ++v) c[v] = 0.0f;

#pragma unroll 4
    for (int kc = 0; kc < 128; ++kc) {
        F16x16 aF, bF;
        aF.q[0] = ha[kc * 4 + half];           // K_local {0..7}+8*half
        aF.q[1] = ha[kc * 4 + 2 + half];       // K_local {16..23}+8*half
        bF.q[0] = gb[col * 512 + kc * 4 + half];
        bF.q[1] = gb[col * 512 + kc * 4 + 2 + half];
        c = __builtin_amdgcn_wmma_f32_16x16x32_f16(
                false, aF.v, false, bF.v, (short)0, c, false, false);
    }

    // D layout: VGPR v -> sample row M = v + 8*half, column e = lane%16
    if (col < 8) {
#pragma unroll
        for (int v = 0; v < 8; ++v)
            logits[wave * 128 + (v + 8 * half) * 8 + col] = c[v];
    }
    __syncthreads();

    if (threadIdx.x < 128) {
        int t = threadIdx.x;               // block-local sample
        const float* lg = logits + (t >> 4) * 128 + (t & 15) * 8;
        int best = 0; float bv = lg[0];
#pragma unroll
        for (int e = 1; e < NEXP; ++e)
            if (lg[e] > bv) { bv = lg[e]; best = e; }    // ties -> lowest idx
        idx[blockIdx.x * 128 + t] = best;
    }
}

// ---------------------------------------------------------------------------
// Expert conv via WMMA implicit GEMM.  One block (8 waves) per sample.
// GEMM: M=16 c_out, N=256 positions (16 tiles, 2 per wave), K=160 (5x32).
// LDS tile halo-padded to 18x18 -> branchless B fragments (tap 9 rows are
// nulled by zero A-fragment columns, so tap is just clamped to 8).
// h_out = 0.95 * relu(A*B + bias), f16.
// ---------------------------------------------------------------------------
__global__ __launch_bounds__(256)
void expert_conv_kernel(const _Float16* __restrict__ hin,
                        _Float16* __restrict__ hout,
                        const int* __restrict__ idx,
                        const _Float16* __restrict__ wfrag,
                        const float* __restrict__ eb) {
    __shared__ uint4 hs[18 * 18 * 2];    // [18 y][18 x][16 ch] f16, halo = 0
    const int b   = blockIdx.x;
    const int tid = threadIdx.x;

    // zero halo (68 border positions)
    if (tid < 68) {
        int y, x;
        if      (tid < 18) { y = 0;        x = tid;      }
        else if (tid < 36) { y = 17;       x = tid - 18; }
        else if (tid < 52) { y = tid - 35; x = 0;        }
        else               { y = tid - 51; x = 17;       }
        hs[(y * 18 + x) * 2]     = make_uint4(0u, 0u, 0u, 0u);
        hs[(y * 18 + x) * 2 + 1] = make_uint4(0u, 0u, 0u, 0u);
    }
    // stage sample interior (coalesced b128)
    {
        const uint4* g = (const uint4*)(hin + (size_t)b * 4096);
        int y = 1 + (tid >> 4), x = 1 + (tid & 15);
        hs[(y * 18 + x) * 2]     = g[tid * 2];
        hs[(y * 18 + x) * 2 + 1] = g[tid * 2 + 1];
    }
    __syncthreads();

    const int e    = idx[b];
    const int wave = tid >> 5;
    const int lane = tid & 31;
    const int half = lane >> 4;
    const int ln   = lane & 15;

    // A fragments for selected expert: 5 chunks x 32B per lane (L2-hot)
    F16x16 aF[5];
    const uint4* wq = (const uint4*)(wfrag + (size_t)e * 5 * 32 * 16);
#pragma unroll
    for (int kc = 0; kc < 5; ++kc) {
        aF[kc].q[0] = wq[(kc * 32 + lane) * 2];
        aF[kc].q[1] = wq[(kc * 32 + lane) * 2 + 1];
    }

    // bias -> accumulator init (C/D layout: VGPR v holds M = v + 8*half)
    v8f c0, c1;
    {
        const float* bp = eb + e * HID + 8 * half;
#pragma unroll
        for (int v = 0; v < 8; ++v) { c0[v] = bp[v]; c1[v] = c0[v]; }
    }

    const int py0 = wave, py1 = wave + 8;     // two position tiles per wave
#pragma unroll
    for (int kc = 0; kc < 5; ++kc) {
        int tap = 2 * kc + half;
        if (tap > 8) tap = 8;                 // K-padding rows nulled by A
        int dy = tap / 3, dx = tap % 3;       // halo offset (no -1)
        int np0 = (py0 + dy) * 18 + (ln + dx);
        int np1 = (py1 + dy) * 18 + (ln + dx);
        F16x16 b0, b1;
        b0.q[0] = hs[np0 * 2]; b0.q[1] = hs[np0 * 2 + 1];
        b1.q[0] = hs[np1 * 2]; b1.q[1] = hs[np1 * 2 + 1];
        c0 = __builtin_amdgcn_wmma_f32_16x16x32_f16(
                 false, aF[kc].v, false, b0.v, (short)0, c0, false, false);
        c1 = __builtin_amdgcn_wmma_f32_16x16x32_f16(
                 false, aF[kc].v, false, b1.v, (short)0, c1, false, false);
    }

    // epilogue: relu, *0.95, f16 pack, one b128 store per lane per tile
    F16x8 o0, o1;
#pragma unroll
    for (int v = 0; v < 8; ++v) {
        o0.h[v] = (_Float16)(SCALE * fmaxf(c0[v], 0.0f));
        o1.h[v] = (_Float16)(SCALE * fmaxf(c1[v], 0.0f));
    }
    _Float16* outb = hout + (size_t)b * 4096 + 8 * half;
    *(uint4*)(outb + (size_t)(py0 * 16 + ln) * 16) = o0.q;
    *(uint4*)(outb + (size_t)(py1 * 16 + ln) * 16) = o1.q;
}

// ---------------------------------------------------------------------------
// Last conv 16->8 s2 + relu + maxpool2 + fc1(relu) + fc2.  One block / sample.
// ---------------------------------------------------------------------------
__global__ __launch_bounds__(128)
void last_kernel(const _Float16* __restrict__ hin,
                 const float* __restrict__ lw_g,
                 const float* __restrict__ lb,
                 const float* __restrict__ fc1w,
                 const float* __restrict__ fc1b,
                 const float* __restrict__ fc2w,
                 const float* __restrict__ fc2b,
                 float* __restrict__ out) {
    __shared__ uint4 hs[512];        // 8 KB activations
    __shared__ float lw[1152];       // last_w [8][16][3][3]
    __shared__ float co[512];        // conv out [8][8][8]
    __shared__ float pooled[128];
    __shared__ float f1[16];

    const int b = blockIdx.x;
    const int t = threadIdx.x;

    const uint4* g = (const uint4*)(hin + (size_t)b * 4096);
#pragma unroll
    for (int k = 0; k < 4; ++k) hs[t + 128 * k] = g[t + 128 * k];
    for (int i = t; i < 1152; i += 128) lw[i] = lw_g[i];
    __syncthreads();

    if (t < 64) {                    // conv: 8 channels per position
        const int oy = t >> 3, ox = t & 7;
        float acc[8];
#pragma unroll
        for (int oc = 0; oc < 8; ++oc) acc[oc] = lb[oc];
#pragma unroll
        for (int ky = 0; ky < 3; ++ky) {
            int iy = 2 * oy + ky - 1;
            if ((unsigned)iy >= 16u) continue;
#pragma unroll
            for (int kx = 0; kx < 3; ++kx) {
                int ix = 2 * ox + kx - 1;
                if ((unsigned)ix >= 16u) continue;
                int np = iy * 16 + ix;
                F16x16 u;
                u.q[0] = hs[np * 2];
                u.q[1] = hs[np * 2 + 1];
#pragma unroll
                for (int ch = 0; ch < 16; ++ch) {
                    float hv = (float)u.h[ch];
#pragma unroll
                    for (int oc = 0; oc < 8; ++oc)
                        acc[oc] = fmaf(hv, lw[((oc * 16 + ch) * 3 + ky) * 3 + kx], acc[oc]);
                }
            }
        }
#pragma unroll
        for (int oc = 0; oc < 8; ++oc) co[oc * 64 + t] = fmaxf(acc[oc], 0.0f);
    }
    __syncthreads();

    {                                 // maxpool 2x2 -> pooled[oc*16+py*4+px]
        int oc = t >> 4, r = t & 15, py = r >> 2, px = r & 3;
        float m0 = co[oc * 64 + (2 * py) * 8 + 2 * px];
        float m1 = co[oc * 64 + (2 * py) * 8 + 2 * px + 1];
        float m2 = co[oc * 64 + (2 * py + 1) * 8 + 2 * px];
        float m3 = co[oc * 64 + (2 * py + 1) * 8 + 2 * px + 1];
        pooled[t] = fmaxf(fmaxf(m0, m1), fmaxf(m2, m3));
    }
    __syncthreads();

    if (t < 16) {                     // fc1 128->16, relu
        float s = fc1b[t];
        const float* wr = fc1w + t * 128;
#pragma unroll 8
        for (int k = 0; k < 128; ++k) s = fmaf(pooled[k], wr[k], s);
        f1[t] = fmaxf(s, 0.0f);
    }
    __syncthreads();

    if (t < 10) {                     // fc2 16->10
        float s = fc2b[t];
        const float* wr = fc2w + t * 16;
#pragma unroll
        for (int k = 0; k < 16; ++k) s = fmaf(f1[k], wr[k], s);
        out[(size_t)b * 10 + t] = s;
    }
}

// ---------------------------------------------------------------------------
extern "C" void kernel_launch(void* const* d_in, const int* in_sizes, int n_in,
                              void* d_out, int out_size, void* d_ws, size_t ws_size,
                              hipStream_t stream) {
    const float* x        = (const float*)d_in[0];
    const float* first_w  = (const float*)d_in[1];
    const float* first_b  = (const float*)d_in[2];
    const float* expert_w = (const float*)d_in[3];
    const float* expert_b = (const float*)d_in[4];
    const float* gate_w   = (const float*)d_in[5];
    const float* last_w   = (const float*)d_in[6];
    const float* last_b   = (const float*)d_in[7];
    const float* fc1_w    = (const float*)d_in[8];
    const float* fc1_b    = (const float*)d_in[9];
    const float* fc2_w    = (const float*)d_in[10];
    const float* fc2_b    = (const float*)d_in[11];
    float* out = (float*)d_out;

    char* ws = (char*)d_ws;
    const size_t H_BYTES = (size_t)BATCH * 4096 * sizeof(_Float16);  // 64 MB
    _Float16* hA      = (_Float16*)(ws);
    _Float16* hB      = (_Float16*)(ws + H_BYTES);
    int*      idx     = (int*)     (ws + 2 * H_BYTES);                // 32 KB
    _Float16* gate_f  = (_Float16*)(ws + 2 * H_BYTES + 65536);        // 128 KB
    _Float16* wfrag   = (_Float16*)(ws + 2 * H_BYTES + 65536 + 131072);

    prep_wfrag_kernel<<<4, 256, 0, stream>>>(expert_w, wfrag);
    prep_gate_kernel<<<256, 256, 0, stream>>>(gate_w, gate_f);
    first_conv_kernel<<<BATCH, 256, 0, stream>>>(x, first_w, first_b, hA);

    _Float16* hc = hA;
    _Float16* hn = hB;
    for (int it = 0; it < 4; ++it) {
        gate_kernel<<<BATCH / 128, 256, 0, stream>>>(hc, gate_f, idx);
        expert_conv_kernel<<<BATCH, 256, 0, stream>>>(hc, hn, idx, wfrag, expert_b);
        _Float16* tmp = hc; hc = hn; hn = tmp;
    }
    last_kernel<<<BATCH, 128, 0, stream>>>(hc, last_w, last_b,
                                           fc1_w, fc1_b, fc2_w, fc2_b, out);
}
